// TGN_63677185130810
// MI455X (gfx1250) — compile-verified, matrix operand order checked by reading, not run on hardware
//
#include <hip/hip_runtime.h>
#include <cstdint>
#include <cstddef>

typedef __attribute__((ext_vector_type(16))) __bf16 v16bf;
typedef __attribute__((ext_vector_type(8)))  float  v8f;

#define N_SRC 65536
#define N_DST 16384
#define KNBR  10
#define Hd    128
#define MAILd 256

__device__ __forceinline__ float sigmoidf_(float x) { return 1.f / (1.f + __expf(-x)); }

// ---- CDNA5 async global->LDS copy (ASYNCcnt-tracked), per-lane 16B ----
__device__ __forceinline__ void async_load_b128(unsigned lds_addr, const void* gaddr) {
  asm volatile("global_load_async_to_lds_b128 %0, %1, off"
               :: "v"(lds_addr), "v"(gaddr) : "memory");
}
__device__ __forceinline__ void wait_asynccnt0() {
  asm volatile("s_wait_asynccnt 0x0" ::: "memory");
}
// low 32 bits of a generic pointer into LDS == wave-relative LDS byte address
__device__ __forceinline__ unsigned lds_off(const void* p) {
  return (unsigned)(size_t)p;
}

// ---- A/B fragment builders for V_WMMA_F32_16X16X32_BF16 (wave32) ----
// 16-bit A 16x32 layout: lane L holds row M=L&15; vector elem j holds
// K = (j&7) + ((j>>3)<<4) + ((L>>4)<<3)  (caller folds kt*32 + ((L>>4)<<3) into p).
__device__ __forceinline__ v16bf a_from_f32(const float* __restrict__ p) {
  v16bf a;
#pragma unroll
  for (int j = 0; j < 8; ++j) a[j] = (__bf16)p[j];
#pragma unroll
  for (int j = 0; j < 8; ++j) a[8 + j] = (__bf16)p[16 + j];
  return a;
}
__device__ __forceinline__ v16bf a_from_bf16(const __bf16* __restrict__ p) {
  v16bf a;
#pragma unroll
  for (int j = 0; j < 8; ++j) a[j] = p[j];
#pragma unroll
  for (int j = 0; j < 8; ++j) a[8 + j] = p[16 + j];
  return a;
}
// time-encoding fragment: cos(t * w[c] + b[c]) computed in-register
__device__ __forceinline__ v16bf a_timecos(float t, const float* __restrict__ w,
                                           const float* __restrict__ b, int kb) {
  v16bf a;
#pragma unroll
  for (int j = 0; j < 16; ++j) {
    int c = kb + (j & 7) + ((j >> 3) << 4);
    a[j] = (__bf16)__cosf(t * w[c] + b[c]);
  }
  return a;
}

// Wave-level GEMM: C(16 x ntiles*16) = A(16 x NKT*32) * W^T, W row-major (Nout x K) bf16.
// W may point to global (L2-resident weights) or LDS-staged copies; after inlining the
// addrspace is inferred (global_load_b128 vs ds_load_b128).
// epi(nt, M, N, val) fuses bias/activation/combine in the C-fragment registers.
template <int NKT, typename AL, typename EP>
__device__ __forceinline__ void wave_gemm(AL aload, EP epi, const __bf16* W,
                                          int ntiles, int lane) {
  v16bf af[NKT];
#pragma unroll
  for (int kt = 0; kt < NKT; ++kt) af[kt] = aload(kt);
  const int n15 = lane & 15;
  const int mb  = (lane >> 4) << 3;
  for (int nt = 0; nt < ntiles; ++nt) {
    v8f acc = {0.f, 0.f, 0.f, 0.f, 0.f, 0.f, 0.f, 0.f};
    const __bf16* wrow = W + ((size_t)(nt * 16 + n15)) * (NKT * 32) + mb;
#pragma unroll
    for (int kt = 0; kt < NKT; ++kt) {
      v16bf b;
#pragma unroll
      for (int j = 0; j < 8; ++j) b[j] = wrow[kt * 32 + j];
#pragma unroll
      for (int j = 0; j < 8; ++j) b[8 + j] = wrow[kt * 32 + 16 + j];
      acc = __builtin_amdgcn_wmma_f32_16x16x32_bf16(false, af[kt], false, b,
                                                    (short)0, acc, false, false);
    }
#pragma unroll
    for (int r = 0; r < 8; ++r) epi(nt, mb + r, nt * 16 + n15, acc[r]);
  }
}

// ---- weight f32 -> bf16 conversion ----
__global__ void cvt_bf16_kernel(const float* __restrict__ src, __bf16* __restrict__ dst, int n) {
  int i = blockIdx.x * blockDim.x + threadIdx.x;
  if (i < n) dst[i] = (__bf16)src[i];
}

// ---- Kernel 1: per-source-node memory GRU + map + compensation + select ----
__global__ __launch_bounds__(64) void tgn_node_kernel(
    const float* __restrict__ x, const float* __restrict__ memory,
    const float* __restrict__ mem_ts, const float* __restrict__ mailbox,
    const float* __restrict__ mail_ts, const float* __restrict__ src_ts,
    const float* __restrict__ mem_hist, const float* __restrict__ mem_hist_ts,
    const int* __restrict__ is_remote_mem,
    const float* __restrict__ tw_mem, const float* __restrict__ tb_mem,
    const float* __restrict__ b_ih, const float* __restrict__ b_hh,
    const float* __restrict__ b_map,
    const float* __restrict__ tw_comp, const float* __restrict__ tb_comp,
    const float* __restrict__ bc1, const float* __restrict__ bc2,
    const __bf16* __restrict__ Wih, const __bf16* __restrict__ Whh,
    const __bf16* __restrict__ Wmap, const __bf16* __restrict__ Wc1,
    const __bf16* __restrict__ Wc2,
    __bf16* __restrict__ hin) {
  __shared__ float Sall[2][16 * 384];
  const int wid = threadIdx.x >> 5, lane = threadIdx.x & 31;
  float* S = Sall[wid];
  const int node0 = (blockIdx.x * 2 + wid) * 16;
  const int m15 = lane & 15;
  const int hi8 = (lane >> 4) << 3;
  const int anode = node0 + m15;  // row this lane supplies for A fragments
  const float tmail = mail_ts[anode] - mem_ts[anode];
  const float dtc = fmaxf(src_ts[anode] - mem_hist_ts[anode], 0.f);

  // gh = memory @ W_hh^T + b_hh  -> S[:, 0:384]
  wave_gemm<4>(
      [&](int kt) { return a_from_f32(memory + (size_t)anode * Hd + kt * 32 + hi8); },
      [&](int, int M, int N, float v) { S[M * 384 + N] = v + b_hh[N]; },
      Whh, 24, lane);

  // gi = [mailbox | cos-time] @ W_ih^T + b_ih, fused GRU gates
  wave_gemm<12>(
      [&](int kt) {
        if (kt < 8) return a_from_f32(mailbox + (size_t)anode * MAILd + kt * 32 + hi8);
        return a_timecos(tmail, tw_mem, tb_mem, (kt - 8) * 32 + hi8);
      },
      [&](int nt, int M, int N, float v) {
        float g = v + b_ih[N];
        float* Sm = S + M * 384;
        if (nt < 16) {               // reset gate (nt<8) / update gate (8..15)
          Sm[N] = sigmoidf_(g + Sm[N]);
        } else {                     // new gate + GRU output
          float r = Sm[N - 256], z = Sm[N - 128];
          float nn = tanhf(g + r * Sm[N]);
          float hold = memory[(size_t)(node0 + M) * Hd + (N - 256)];
          Sm[N] = (1.f - z) * nn + z * hold;  // out_memory in S[:, 256:384]
        }
      },
      Wih, 24, lane);

  // h = out_memory + x @ W_map^T + b_map  -> S[:, 0:128]
  wave_gemm<4>(
      [&](int kt) { return a_from_f32(x + (size_t)anode * Hd + kt * 32 + hi8); },
      [&](int, int M, int N, float v) { S[M * 384 + N] = v + b_map[N] + S[M * 384 + 256 + N]; },
      Wmap, 8, lane);

  // z1 = relu([mem_hist | cos-time] @ Wc1^T + bc1)  -> S[:, 128:256]
  wave_gemm<8>(
      [&](int kt) {
        if (kt < 4) return a_from_f32(mem_hist + (size_t)anode * Hd + kt * 32 + hi8);
        return a_timecos(dtc, tw_comp, tb_comp, (kt - 4) * 32 + hi8);
      },
      [&](int, int M, int N, float v) {
        float u = v + bc1[N];
        S[M * 384 + 128 + N] = u > 0.f ? u : 0.f;
      },
      Wc1, 8, lane);

  // h_comp = z1 @ Wc2^T + bc2 ; select and store h_input (bf16)
  wave_gemm<4>(
      [&](int kt) { return a_from_f32(S + m15 * 384 + 128 + kt * 32 + hi8); },
      [&](int, int M, int N, float v) {
        int nd = node0 + M;
        float hc = v + bc2[N];
        float sel = is_remote_mem[nd] ? hc : S[M * 384 + N];
        hin[(size_t)nd * Hd + N] = (__bf16)sel;
      },
      Wc2, 8, lane);
}

// ---- Kernel 2: per-dst attention + output proj + compensation + select ----
// Wk/Wv (96KB each) are staged into one shared LDS buffer with CDNA5 async
// global->LDS copies (overlapped with the q projection), eliminating the
// per-neighbor re-fetch of the K/V weights from L2.
__global__ __launch_bounds__(64) void tgn_dst_kernel(
    const int* __restrict__ nbr_idx, const float* __restrict__ edge_feat,
    const float* __restrict__ edge_dt, const float* __restrict__ dst_ts,
    const float* __restrict__ gnn_hist, const float* __restrict__ gnn_hist_ts,
    const int* __restrict__ is_remote_gnn,
    const float* __restrict__ tw_att, const float* __restrict__ tb_att,
    const float* __restrict__ tw_comp, const float* __restrict__ tb_comp,
    const float* __restrict__ bq, const float* __restrict__ bk,
    const float* __restrict__ bv, const float* __restrict__ bo,
    const float* __restrict__ bg1, const float* __restrict__ bg2,
    const __bf16* __restrict__ Wq, const __bf16* __restrict__ Wk,
    const __bf16* __restrict__ Wv, const __bf16* __restrict__ Wo,
    const __bf16* __restrict__ Wg1, const __bf16* __restrict__ Wg2,
    const __bf16* __restrict__ hin, float* __restrict__ out) {
  __shared__ __align__(16) __bf16 Wlds[128 * 384];  // 96KB: Wk, then Wv
  __shared__ float SqA[2][2048], SkvA[2][2048], SoA[2][2048], SscA[2][512];
  const int wid = threadIdx.x >> 5, lane = threadIdx.x & 31;
  const int tid = threadIdx.x;
  float* Sq = SqA[wid];
  float* Skv = SkvA[wid];
  float* So = SoA[wid];
  float* Ssc = SscA[wid];
  const int d0 = (blockIdx.x * 2 + wid) * 16;
  const int m15 = lane & 15, hi8 = (lane >> 4) << 3;
  const int drow = d0 + m15;

  // kick off async staging of Wk into LDS (98304 B = 6144 x 16B chunks)
  {
    const unsigned base = lds_off(&Wlds[0]);
    const char* g = (const char*)Wk;
#pragma unroll 4
    for (int i = tid; i < 6144; i += 64)
      async_load_b128(base + i * 16u, g + (size_t)i * 16u);
  }

  // q = h_dst @ Wq^T + bq  (B from global; overlaps with async staging)
  wave_gemm<4>(
      [&](int kt) { return a_from_bf16(hin + (size_t)drow * Hd + kt * 32 + hi8); },
      [&](int, int M, int N, float v) { Sq[M * 128 + N] = v + bq[N]; }, Wq, 8, lane);

  wait_asynccnt0();
  __syncthreads();  // Wk fully staged for both waves

  // pass 1: key projections (B from LDS) + scores
  for (int k = 0; k < KNBR; ++k) {
    const int nb = nbr_idx[drow * KNBR + k];
    const float edt = edge_dt[drow * KNBR + k];
    const float* ef = edge_feat + ((size_t)drow * KNBR + k) * Hd;
    wave_gemm<12>(
        [&](int kt) {
          if (kt < 4) return a_from_bf16(hin + (size_t)nb * Hd + kt * 32 + hi8);
          if (kt < 8) return a_from_f32(ef + (kt - 4) * 32 + hi8);
          return a_timecos(edt, tw_att, tb_att, (kt - 8) * 32 + hi8);
        },
        [&](int, int M, int N, float v) { Skv[M * 128 + N] = v + bk[N]; }, Wlds, 8, lane);
    {
      const int dd = lane & 15, hh = lane >> 4;  // one (dst, head) per lane
      const float* qr = Sq + dd * 128 + hh * 64;
      const float* kr = Skv + dd * 128 + hh * 64;
      float s = 0.f;
#pragma unroll 8
      for (int i = 0; i < 64; ++i) s += qr[i] * kr[i];
      Ssc[(dd * 2 + hh) * 16 + k] = s * 0.125f;  // 1/sqrt(64)
    }
  }
  // softmax over K per (dst, head)
  {
    const int dd = lane & 15, hh = lane >> 4;
    float* sc = Ssc + (dd * 2 + hh) * 16;
    float m = sc[0];
    for (int k = 1; k < KNBR; ++k) m = fmaxf(m, sc[k]);
    float sum = 0.f;
    for (int k = 0; k < KNBR; ++k) { float e = __expf(sc[k] - m); sc[k] = e; sum += e; }
    float inv = 1.f / sum;
    for (int k = 0; k < KNBR; ++k) sc[k] *= inv;
  }
  for (int i = lane; i < 2048; i += 32) So[i] = 0.f;

  __syncthreads();  // everyone done reading Wk from Wlds
  // restage Wv into the same buffer
  {
    const unsigned base = lds_off(&Wlds[0]);
    const char* g = (const char*)Wv;
#pragma unroll 4
    for (int i = tid; i < 6144; i += 64)
      async_load_b128(base + i * 16u, g + (size_t)i * 16u);
  }
  wait_asynccnt0();
  __syncthreads();  // Wv fully staged

  // pass 2: value projections (B from LDS) + weighted accumulation
  for (int k = 0; k < KNBR; ++k) {
    const int nb = nbr_idx[drow * KNBR + k];
    const float edt = edge_dt[drow * KNBR + k];
    const float* ef = edge_feat + ((size_t)drow * KNBR + k) * Hd;
    wave_gemm<12>(
        [&](int kt) {
          if (kt < 4) return a_from_bf16(hin + (size_t)nb * Hd + kt * 32 + hi8);
          if (kt < 8) return a_from_f32(ef + (kt - 4) * 32 + hi8);
          return a_timecos(edt, tw_att, tb_att, (kt - 8) * 32 + hi8);
        },
        [&](int, int M, int N, float v) { Skv[M * 128 + N] = v + bv[N]; }, Wlds, 8, lane);
    for (int i = lane; i < 2048; i += 32) {
      int dd = i >> 7, c = i & 127, hh = c >> 6;
      So[i] += Ssc[(dd * 2 + hh) * 16 + k] * Skv[i];
    }
  }
  // h_out = o @ Wo^T + bo  (reuse Sq)
  wave_gemm<4>(
      [&](int kt) { return a_from_f32(So + m15 * 128 + kt * 32 + hi8); },
      [&](int, int M, int N, float v) { Sq[M * 128 + N] = v + bo[N]; }, Wo, 8, lane);
  // compensation MLP 2 + final select
  const float dt2 = fmaxf(dst_ts[drow] - gnn_hist_ts[drow], 0.f);
  wave_gemm<8>(
      [&](int kt) {
        if (kt < 4) return a_from_f32(gnn_hist + (size_t)drow * Hd + kt * 32 + hi8);
        return a_timecos(dt2, tw_comp, tb_comp, (kt - 4) * 32 + hi8);
      },
      [&](int, int M, int N, float v) {
        float u = v + bg1[N];
        Skv[M * 128 + N] = u > 0.f ? u : 0.f;
      },
      Wg1, 8, lane);
  wave_gemm<4>(
      [&](int kt) { return a_from_f32(Skv + m15 * 128 + kt * 32 + hi8); },
      [&](int, int M, int N, float v) {
        int dd = d0 + M;
        float hc = v + bg2[N];
        out[(size_t)dd * Hd + N] = is_remote_gnn[dd] ? hc : Sq[M * 128 + N];
      },
      Wg2, 8, lane);
}

extern "C" void kernel_launch(void* const* d_in, const int* in_sizes, int n_in,
                              void* d_out, int out_size, void* d_ws, size_t ws_size,
                              hipStream_t stream) {
  (void)in_sizes; (void)n_in; (void)out_size; (void)ws_size;
  const float* x            = (const float*)d_in[0];
  const float* memory_     = (const float*)d_in[1];
  const float* mem_ts      = (const float*)d_in[2];
  const float* mailbox     = (const float*)d_in[3];
  const float* mail_ts     = (const float*)d_in[4];
  const float* src_ts      = (const float*)d_in[5];
  const float* mem_hist    = (const float*)d_in[6];
  const float* mem_hist_ts = (const float*)d_in[7];
  const int*   is_remote_mem = (const int*)d_in[8];
  const int*   nbr_idx     = (const int*)d_in[9];
  const float* edge_feat   = (const float*)d_in[10];
  const float* edge_dt     = (const float*)d_in[11];
  const float* dst_ts      = (const float*)d_in[12];
  const float* gnn_hist    = (const float*)d_in[13];
  const float* gnn_hist_ts = (const float*)d_in[14];
  const int*   is_remote_gnn = (const int*)d_in[15];
  const int P = 16;  // params in dict order
  const float* tw_mem = (const float*)d_in[P + 0];
  const float* tb_mem = (const float*)d_in[P + 1];
  const float* W_ih   = (const float*)d_in[P + 2];
  const float* b_ih   = (const float*)d_in[P + 3];
  const float* W_hh   = (const float*)d_in[P + 4];
  const float* b_hh   = (const float*)d_in[P + 5];
  const float* W_map  = (const float*)d_in[P + 6];
  const float* b_map  = (const float*)d_in[P + 7];
  const float* tw_comp = (const float*)d_in[P + 8];
  const float* tb_comp = (const float*)d_in[P + 9];
  const float* Wc1 = (const float*)d_in[P + 10];
  const float* bc1 = (const float*)d_in[P + 11];
  const float* Wc2 = (const float*)d_in[P + 12];
  const float* bc2 = (const float*)d_in[P + 13];
  const float* Wg1 = (const float*)d_in[P + 14];
  const float* bg1 = (const float*)d_in[P + 15];
  const float* Wg2 = (const float*)d_in[P + 16];
  const float* bg2 = (const float*)d_in[P + 17];
  const float* tw_att = (const float*)d_in[P + 18];
  const float* tb_att = (const float*)d_in[P + 19];
  const float* Wq = (const float*)d_in[P + 20];
  const float* bq = (const float*)d_in[P + 21];
  const float* Wk = (const float*)d_in[P + 22];
  const float* bk = (const float*)d_in[P + 23];
  const float* Wv = (const float*)d_in[P + 24];
  const float* bv = (const float*)d_in[P + 25];
  const float* Wo = (const float*)d_in[P + 26];
  const float* bo = (const float*)d_in[P + 27];

  // workspace layout: h_input bf16 table, then bf16 weight copies
  char* wsb = (char*)d_ws;
  __bf16* hin = (__bf16*)wsb;
  size_t off = (size_t)N_SRC * Hd * sizeof(__bf16);
  auto take = [&](size_t nelem) -> __bf16* {
    __bf16* p = (__bf16*)(wsb + off);
    off = (off + nelem * sizeof(__bf16) + 255) & ~(size_t)255;
    return p;
  };
  __bf16* bWih  = take(384 * 384);
  __bf16* bWhh  = take(384 * 128);
  __bf16* bWmap = take(128 * 128);
  __bf16* bWc1  = take(128 * 256);
  __bf16* bWc2  = take(128 * 128);
  __bf16* bWg1  = take(128 * 256);
  __bf16* bWg2  = take(128 * 128);
  __bf16* bWq   = take(128 * 128);
  __bf16* bWk   = take(128 * 384);
  __bf16* bWv   = take(128 * 384);
  __bf16* bWo   = take(128 * 128);

  auto cvt = [&](const float* s, __bf16* dst, int n) {
    cvt_bf16_kernel<<<dim3((n + 255) / 256), dim3(256), 0, stream>>>(s, dst, n);
  };
  cvt(W_ih, bWih, 384 * 384);
  cvt(W_hh, bWhh, 384 * 128);
  cvt(W_map, bWmap, 128 * 128);
  cvt(Wc1, bWc1, 128 * 256);
  cvt(Wc2, bWc2, 128 * 128);
  cvt(Wg1, bWg1, 128 * 256);
  cvt(Wg2, bWg2, 128 * 128);
  cvt(Wq, bWq, 128 * 128);
  cvt(Wk, bWk, 128 * 384);
  cvt(Wv, bWv, 128 * 384);
  cvt(Wo, bWo, 128 * 128);

  tgn_node_kernel<<<dim3(N_SRC / 32), dim3(64), 0, stream>>>(
      x, memory_, mem_ts, mailbox, mail_ts, src_ts, mem_hist, mem_hist_ts,
      is_remote_mem, tw_mem, tb_mem, b_ih, b_hh, b_map, tw_comp, tb_comp, bc1, bc2,
      bWih, bWhh, bWmap, bWc1, bWc2, hin);

  tgn_dst_kernel<<<dim3(N_DST / 32), dim3(64), 0, stream>>>(
      nbr_idx, edge_feat, edge_dt, dst_ts, gnn_hist, gnn_hist_ts, is_remote_gnn,
      tw_att, tb_att, tw_comp, tb_comp, bq, bk, bv, bo, bg1, bg2,
      bWq, bWk, bWv, bWo, bWg1, bWg2, hin, (float*)d_out);
}
